// PhysicsEdgeBias_5437428597336
// MI455X (gfx1250) — matrix operand order repaired and checked by reading, not compiled, not run on hardware
//
#include <hip/hip_runtime.h>
#include <hip/hip_bf16.h>

typedef _Float16 f16;
typedef __attribute__((ext_vector_type(16))) _Float16 v16h;
typedef __attribute__((ext_vector_type(8)))  _Float16 v8h;
typedef __attribute__((ext_vector_type(8)))  float    v8f;

#define TAU   64
#define EMB   8
#define HID   32
#define DIN   16
#define LSZ   1024

// Workspace layout (f16):
//   ai16 : [B*L][32]  @ 0        (262144 B)  (x @ W1[:16],  query/i part)
//   aj16 : [B*L][32]  @ 262144   (262144 B)  (x @ W1[16:32], key/j part)
//   ae16 : [65][32]   @ 524288   (4160 B)    (dt_table @ W1[32:40] + b1)
//   w2h  : [32]       @ 528448   (64 B)      (W2 in f16)

// ---------------- Stage 1a: ai / aj projections (trivial GEMMs, ~4 MFLOP) ----
__global__ __launch_bounds__(256) void k_proj(const float* __restrict__ x,
                                              const float* __restrict__ W1,
                                              f16* __restrict__ ai16,
                                              f16* __restrict__ aj16) {
  int idx = blockIdx.x * 256 + threadIdx.x;   // [0, 4096*32)
  int r = idx >> 5;
  int h = idx & 31;
  const float* xr = x + r * DIN;
  float sa = 0.f, sb = 0.f;
#pragma unroll
  for (int d = 0; d < DIN; ++d) {
    float xv = xr[d];
    sa = fmaf(xv, W1[d * HID + h],          sa);
    sb = fmaf(xv, W1[(DIN + d) * HID + h],  sb);
  }
  ai16[idx] = (f16)sa;
  aj16[idx] = (f16)sb;
}

// ---------------- Stage 1b: ae table (65x32, b1 folded) + W2 in f16 ---------
__global__ __launch_bounds__(256) void k_emb(const float* __restrict__ dt_table,
                                             const float* __restrict__ W1,
                                             const float* __restrict__ b1,
                                             const float* __restrict__ W2,
                                             f16* __restrict__ ae16,
                                             f16* __restrict__ w2h) {
  int idx = blockIdx.x * 256 + threadIdx.x;
  if (idx < (TAU + 1) * HID) {
    int t = idx >> 5;
    int h = idx & 31;
    float s = b1[h];
#pragma unroll
    for (int e = 0; e < EMB; ++e)
      s = fmaf(dt_table[t * EMB + e], W1[(2 * DIN + e) * HID + h], s);
    ae16[idx] = (f16)s;
  }
  if (idx < HID) w2h[idx] = (f16)W2[idx];
}

// ---------------- Stage 2: main bias kernel (WMMA reduction over H=32) ------
// Block = one (b, i) output row (4096 blocks), 4 waves/block.
// Each wave owns 256 consecutive j's = 16 WMMA tiles of 16 j's, processed
// two-at-a-time with independent accumulators so the WMMA->VALU D-read
// hazard (4 coexec NOPs for *F16 WMMA) is always covered by real work and
// nothing is carried across the loop backedge (no register copies).
__global__ __launch_bounds__(128) void k_bias(const f16* __restrict__ ai16,
                                              const f16* __restrict__ aj16,
                                              const f16* __restrict__ ae16,
                                              const f16* __restrict__ w2h,
                                              const float* __restrict__ b2p,
                                              float* __restrict__ out) {
  const int lane = threadIdx.x & 31;
  const int wave = threadIdx.x >> 5;          // 0..3
  const int M    = lane & 15;                 // row within WMMA tile
  const int g    = lane >> 4;                 // lane group (selects K chunks)
  const int bi   = blockIdx.x;                // b*L + i
  const int i    = bi & (LSZ - 1);
  const float b2 = b2p[0];
  float* out_row = out + (size_t)bi * LSZ;

  // ai row is constant over all j handled by this block; hoist its two
  // 16-byte K-chunks (K in [8g,8g+8) and [16+8g,16+8g+8), per A layout).
  const v8h ai_lo = *(const v8h*)(ai16 + bi * HID + 8 * g);
  const v8h ai_hi = *(const v8h*)(ai16 + bi * HID + 16 + 8 * g);

  // B matrix: lane group g holds K in [8g,8g+8) and [16+8g,16+8g+8) (same as A)
  const v8h blo = *(const v8h*)(w2h + 8 * g);
  const v8h bhi = *(const v8h*)(w2h + 16 + 8 * g);
  const v16h bmat = __builtin_shufflevector(blo, bhi,
      0, 1, 2, 3, 4, 5, 6, 7, 8, 9, 10, 11, 12, 13, 14, 15);

  const f16* ajb = aj16 + (size_t)(bi - i) * HID;   // start of batch b rows
  const int jwave = wave * 256;
  const v8h zero8 = {};

  // Build the 16x32 f16 A tile (relu'd pre-activations) for tile base j0.
  auto buildA = [&](int j0) -> v16h {
    const int j = j0 + M;                           // this lane's A-row pair
    int d = i - j;
    d = d < 0 ? 0 : (d > TAU ? TAU : d);
    const f16* ajr = ajb + j * HID;
    const f16* aer = ae16 + d * HID;
    __builtin_prefetch(ajr + 32 * HID, 0, 3);       // aj rows two tiles ahead
    v8h alo = ai_lo + *(const v8h*)(ajr + 8 * g) + *(const v8h*)(aer + 8 * g);
    v8h ahi = ai_hi + *(const v8h*)(ajr + 16 + 8 * g) + *(const v8h*)(aer + 16 + 8 * g);
    alo = __builtin_elementwise_max(alo, zero8);
    ahi = __builtin_elementwise_max(ahi, zero8);
    return __builtin_shufflevector(alo, ahi,
        0, 1, 2, 3, 4, 5, 6, 7, 8, 9, 10, 11, 12, 13, 14, 15);
  };

  // D layout: c[v] holds output row (v + 8g) at lane group g, identical in
  // every N column. Lane (M, g) with M < 8 owns row M + 8g and extracts
  // c[M] via a branch-free cndmask select tree, then one coalesced store
  // (16 active lanes -> 16 consecutive floats, one cacheline).
  auto epilogue = [&](v8f c, int j0) {
    const bool m1 = (M & 1) != 0;
    const bool m2 = (M & 2) != 0;
    const bool m4 = (M & 4) != 0;
    float t0 = m1 ? c[1] : c[0];
    float t1 = m1 ? c[3] : c[2];
    float t2 = m1 ? c[5] : c[4];
    float t3 = m1 ? c[7] : c[6];
    float u0 = m2 ? t1 : t0;
    float u1 = m2 ? t3 : t2;
    float val = m4 ? u1 : u0;
    const int jo = j0 + M + 8 * g;                  // row this lane stores
    int dd = i - jo;
    dd = dd < 0 ? 0 : (dd > TAU ? TAU : dd);        // -> v_med3_i32
    val = val + b2 - 0.2f * (float)dd;
    if (M < 8) out_row[jo] = val;
  };

  // Two independent tiles per iteration; epilogues run after both WMMAs so
  // each D-read has >= 13 independent VALU ops since its producing WMMA.
  for (int t = 0; t < 16; t += 2) {
    const int j0 = jwave + t * 16;

    v16h a0 = buildA(j0);
    v8f c0 = {};
    c0 = __builtin_amdgcn_wmma_f32_16x16x32_f16(false, a0, false, bmat,
                                                (short)0, c0, false, false);

    v16h a1 = buildA(j0 + 16);
    v8f c1 = {};
    c1 = __builtin_amdgcn_wmma_f32_16x16x32_f16(false, a1, false, bmat,
                                                (short)0, c1, false, false);

    epilogue(c0, j0);
    epilogue(c1, j0 + 16);
  }
}

extern "C" void kernel_launch(void* const* d_in, const int* in_sizes, int n_in,
                              void* d_out, int out_size, void* d_ws, size_t ws_size,
                              hipStream_t stream) {
  const float* x        = (const float*)d_in[0];   // [4,1024,16]
  const float* dt_table = (const float*)d_in[1];   // [65,8]
  const float* W1       = (const float*)d_in[2];   // [40,32]
  const float* b1       = (const float*)d_in[3];   // [32]
  const float* W2       = (const float*)d_in[4];   // [32]
  const float* b2       = (const float*)d_in[5];   // scalar
  float* out = (float*)d_out;                      // [4,1024,1024]

  char* ws = (char*)d_ws;
  f16* ai16 = (f16*)(ws);
  f16* aj16 = (f16*)(ws + 262144);
  f16* ae16 = (f16*)(ws + 524288);
  f16* w2h  = (f16*)(ws + 528448);

  k_proj<<<512, 256, 0, stream>>>(x, W1, ai16, aj16);
  k_emb<<<9, 256, 0, stream>>>(dt_table, W1, b1, W2, ae16, w2h);
  k_bias<<<4096, 128, 0, stream>>>(ai16, aj16, ae16, w2h, b2, out);
}
